// TransformerEncoder_77318001262623
// MI455X (gfx1250) — compile-verified, matrix operand order checked by reading, not run on hardware
//
#include <hip/hip_runtime.h>
#include <hip/hip_bf16.h>
#include <stdint.h>
#include <stddef.h>

// ---------------- problem constants (match reference) ----------------
constexpr int Lc = 6, Dc = 512, Fc = 2048, Hc = 8, DHc = 64, Bc = 2, Sc = 2048;
constexpr int NT = Bc * Sc;          // 4096 token rows
constexpr float SCL = 0.125f;        // DH^-0.5

// ---------------- CDNA5 WMMA types ----------------
typedef __attribute__((ext_vector_type(16))) __bf16 v16bf;
typedef __attribute__((ext_vector_type(8)))  __bf16 v8bf;
typedef __attribute__((ext_vector_type(8)))  float  v8f;

__device__ __forceinline__ unsigned short f2bf(float f) {
  unsigned u = __float_as_uint(f);
  return (unsigned short)((u + 0x7FFFu + ((u >> 16) & 1u)) >> 16);
}

__device__ __forceinline__ v8f vzero() {
  v8f z;
#pragma unroll
  for (int i = 0; i < 8; i++) z[i] = 0.f;
  return z;
}

// Fragment = two contiguous 16B halves (A-layout: K = i + 8*(lane/16), +8 for i>=8)
__device__ __forceinline__ v16bf ldfrag(const unsigned short* lo, const unsigned short* hi) {
  v8bf a = *(const v8bf*)lo;
  v8bf b = *(const v8bf*)hi;
  return __builtin_shufflevector(a, b, 0, 1, 2, 3, 4, 5, 6, 7, 8, 9, 10, 11, 12, 13, 14, 15);
}

__device__ __forceinline__ v8f wmma_bf16(v16bf a, v16bf b, v8f c) {
  return __builtin_amdgcn_wmma_f32_16x16x32_bf16(false, a, false, b, (short)0, c, false, false);
}

// ---------------- fp32 -> bf16 weight conversion WITH transpose ----------------
// src: [nmat, K, N] fp32 row-major  ->  dst: [nmat, N, K] bf16 row-major
// One-time cost per launch; makes every GEMM B-tile stage & fragment load fully
// vectorized (ds_store_b128 / ds_load_b128) instead of 16-bit scatter stores.
__global__ void cvt_t_kernel(const float* __restrict__ src, unsigned short* __restrict__ dst,
                             int K, int N, size_t total) {
  size_t i = (size_t)blockIdx.x * blockDim.x + threadIdx.x;
  size_t stride = (size_t)gridDim.x * blockDim.x;
  const size_t mat = (size_t)K * N;
  for (; i < total; i += stride) {
    size_t l = i / mat, r = i - l * mat;
    size_t k = r / N, n = r - k * N;
    dst[l * mat + n * (size_t)K + k] = f2bf(src[i]);
  }
}

// ---------------- LayerNorm (row = 512, block = 256 threads) ----------------
__device__ __forceinline__ float blockSum256(float v) {
  __shared__ float red[8];
#pragma unroll
  for (int o = 16; o > 0; o >>= 1) v += __shfl_xor(v, o, 32);
  if ((threadIdx.x & 31) == 0) red[threadIdx.x >> 5] = v;
  __syncthreads();
  float s = red[0];
#pragma unroll
  for (int i = 1; i < 8; i++) s += red[i];
  __syncthreads();
  return s;
}

__global__ __launch_bounds__(256) void ln_kernel(
    const float* __restrict__ x, const float* __restrict__ w, const float* __restrict__ b,
    float* __restrict__ outF, unsigned short* __restrict__ outB) {
  const int row = blockIdx.x, tid = threadIdx.x;
  const float v0 = x[(size_t)row * Dc + tid];
  const float v1 = x[(size_t)row * Dc + tid + 256];
  const float mean = blockSum256(v0 + v1) * (1.f / Dc);
  const float d0 = v0 - mean, d1 = v1 - mean;
  const float var = blockSum256(d0 * d0 + d1 * d1) * (1.f / Dc);
  const float rs = rsqrtf(var + 1e-5f);
  const float y0 = d0 * rs * w[tid] + b[tid];
  const float y1 = d1 * rs * w[tid + 256] + b[tid + 256];
  if (outF) {
    outF[(size_t)row * Dc + tid] = y0;
    outF[(size_t)row * Dc + tid + 256] = y1;
  } else {
    outB[(size_t)row * Dc + tid] = f2bf(y0);
    outB[(size_t)row * Dc + tid + 256] = f2bf(y1);
  }
}

// ---------------- BF16 GEMM: C[M,N] = A[M,K] * W[K,N] (+bias, epilogues) ----------------
// A   : [M,K] bf16 row-major
// Bwt : W TRANSPOSED, [N,K] bf16 row-major  (pre-transposed by cvt_t_kernel)
// mode 0: scatter to Q[B,H,S,DH], K[B,H,S,DH], V^T[B,H,DH,S]  (bf16)
// mode 1: out fp32 = C + bias + res (residual add, in-place with h allowed)
// mode 2: out bf16 = relu(C + bias)
__global__ __launch_bounds__(256) void gemm_bf16(
    const unsigned short* __restrict__ A, const unsigned short* __restrict__ Bwt,
    const float* __restrict__ bias, const float* __restrict__ res,
    float* __restrict__ outF, unsigned short* __restrict__ outB,
    unsigned short* __restrict__ qo, unsigned short* __restrict__ ko,
    unsigned short* __restrict__ vo, int M, int N, int K, int mode) {
  __shared__ alignas(16) unsigned short As[128][40];   // [m][k] (+pad, 80B row = 16B mult)
  __shared__ alignas(16) unsigned short Bs[128][40];   // [n][k] (+pad)

  const int tid = threadIdx.x;
  const int wave = tid >> 5, lane = tid & 31;
  const int g = lane >> 4, tr = lane & 15;
  const int wm = wave >> 2, wn = wave & 3;          // 2x4 wave grid
  const int mBase = blockIdx.y * 128;
  const int nBase = blockIdx.x * 128;
  const int waveM = wm * 64, waveN = wn * 32;

  v8f acc[4][2];
#pragma unroll
  for (int i = 0; i < 4; i++)
#pragma unroll
    for (int j = 0; j < 2; j++) acc[i][j] = vzero();

  const int ar = tid >> 1, ah = tid & 1;   // staging: 2 threads/row, 16 ushorts each

  for (int k0 = 0; k0 < K; k0 += 32) {
    uint4 a0, a1, b0, b1;
    {
      const uint4* ag = (const uint4*)(A + (size_t)(mBase + ar) * K + k0 + ah * 16);
      a0 = ag[0]; a1 = ag[1];
      const uint4* bg = (const uint4*)(Bwt + (size_t)(nBase + ar) * K + k0 + ah * 16);
      b0 = bg[0]; b1 = bg[1];
    }
    __syncthreads();   // previous iteration's fragment reads done
    *(uint4*)&As[ar][ah * 16] = a0;
    *(uint4*)&As[ar][ah * 16 + 8] = a1;
    *(uint4*)&Bs[ar][ah * 16] = b0;
    *(uint4*)&Bs[ar][ah * 16 + 8] = b1;
    __syncthreads();

    v16bf af[4], bf_[2];
#pragma unroll
    for (int mi = 0; mi < 4; mi++) {
      const unsigned short* rp = &As[waveM + mi * 16 + tr][0];
      af[mi] = ldfrag(rp + 8 * g, rp + 16 + 8 * g);
    }
#pragma unroll
    for (int ni = 0; ni < 2; ni++) {
      const unsigned short* cp = &Bs[waveN + ni * 16 + tr][0];
      bf_[ni] = ldfrag(cp + 8 * g, cp + 16 + 8 * g);
    }
#pragma unroll
    for (int mi = 0; mi < 4; mi++)
#pragma unroll
      for (int ni = 0; ni < 2; ni++)
        acc[mi][ni] = wmma_bf16(af[mi], bf_[ni], acc[mi][ni]);
  }

  // epilogue: C lane layout -> M = 8*(lane/16)+vgpr, N = lane%16
#pragma unroll
  for (int mi = 0; mi < 4; mi++) {
#pragma unroll
    for (int ni = 0; ni < 2; ni++) {
#pragma unroll
      for (int r = 0; r < 8; r++) {
        const int m = mBase + waveM + mi * 16 + 8 * g + r;
        const int c = nBase + waveN + ni * 16 + tr;
        float v = acc[mi][ni][r] + bias[c];
        if (mode == 1) {
          outF[(size_t)m * N + c] = v + res[(size_t)m * N + c];
        } else if (mode == 2) {
          outB[(size_t)m * N + c] = f2bf(fmaxf(v, 0.f));
        } else {
          const int bb = m >> 11, ss = m & (Sc - 1);        // Sc = 2048
          const int part = c >> 9, cc = c & (Dc - 1);       // Dc = 512
          const int hh = cc >> 6, dd = cc & (DHc - 1);      // DHc = 64
          if (part == 0)
            qo[(((size_t)bb * Hc + hh) * Sc + ss) * DHc + dd] = f2bf(v);
          else if (part == 1)
            ko[(((size_t)bb * Hc + hh) * Sc + ss) * DHc + dd] = f2bf(v);
          else
            vo[(((size_t)bb * Hc + hh) * DHc + dd) * Sc + ss] = f2bf(v);
        }
      }
    }
  }
}

// ---------------- Attention: one block per (b, h, 16 query rows) ----------------
// LDS: sc[16][Sc] fp32 scores (128KB) | pb[16][Sc] bf16 probs (64KB) | Qs[16][DHc] (2KB)
__global__ __launch_bounds__(256) void attn_kernel(
    const unsigned short* __restrict__ Qb, const unsigned short* __restrict__ Kb,
    const unsigned short* __restrict__ Vt, const float* __restrict__ mask,
    unsigned short* __restrict__ Ob) {
  extern __shared__ char smem[];
  float* sc = (float*)smem;
  unsigned short* pb = (unsigned short*)(smem + (size_t)16 * Sc * 4);
  unsigned short* Qs = (unsigned short*)(smem + (size_t)16 * Sc * 4 + (size_t)16 * Sc * 2);

  const int tid = threadIdx.x;
  const int wave = tid >> 5, lane = tid & 31, g = lane >> 4, tr = lane & 15;
  const int qt = blockIdx.x & (Sc / 16 - 1);
  const int bh = blockIdx.x / (Sc / 16);
  const int bb = bh / Hc, hh = bh % Hc;
  const size_t qkBase = (size_t)bh * Sc * DHc;

  if (tid < 128) {  // 16x64 Q tile
    const int r = tid >> 3, p = tid & 7;
    *(uint4*)&Qs[r * DHc + p * 8] = *(const uint4*)&Qb[qkBase + (size_t)(qt * 16 + r) * DHc + p * 8];
  }
  __syncthreads();

  // phase 1: scores = scale * Q K^T + mask
  for (int kt = wave; kt < Sc / 16; kt += 8) {
    v8f a = vzero();
    const int sk = kt * 16 + tr;
#pragma unroll
    for (int ks = 0; ks < 2; ks++) {
      v16bf af = ldfrag(&Qs[tr * DHc + ks * 32 + 8 * g], &Qs[tr * DHc + ks * 32 + 16 + 8 * g]);
      const unsigned short* kp = &Kb[qkBase + (size_t)sk * DHc + ks * 32];
      v16bf bf_ = ldfrag(kp + 8 * g, kp + 16 + 8 * g);
      a = wmma_bf16(af, bf_, a);
    }
    const float mv = mask[(size_t)bb * Sc + sk];
#pragma unroll
    for (int r = 0; r < 8; r++) sc[(size_t)(8 * g + r) * Sc + sk] = a[r] * SCL + mv;
  }
  __syncthreads();

  // phase 2: exact softmax per row (16 threads per row, wave32-safe 16-lane shuffles)
  {
    const int row = tid >> 4, l16 = tid & 15;
    float mx = -3.4e38f;
    for (int c = l16; c < Sc; c += 16) mx = fmaxf(mx, sc[(size_t)row * Sc + c]);
#pragma unroll
    for (int o = 8; o > 0; o >>= 1) mx = fmaxf(mx, __shfl_xor(mx, o, 16));
    float sum = 0.f;
    for (int c = l16; c < Sc; c += 16) {
      float e = __expf(sc[(size_t)row * Sc + c] - mx);
      sc[(size_t)row * Sc + c] = e;
      sum += e;
    }
#pragma unroll
    for (int o = 8; o > 0; o >>= 1) sum += __shfl_xor(sum, o, 16);
    const float inv = 1.f / sum;
    for (int c = l16; c < Sc; c += 16) pb[(size_t)row * Sc + c] = f2bf(sc[(size_t)row * Sc + c] * inv);
  }
  __syncthreads();

  // phase 3: O = P @ V  (split-K over 2 wave groups, 4 n-tiles of 16)
  const int nt = wave & 3, half = wave >> 2;
  v8f a = vzero();
  const size_t vBase = ((size_t)bh * DHc + nt * 16 + tr) * Sc;
  for (int ks = 0; ks < 32; ks++) {
    const int kk = half * 1024 + ks * 32;
    v16bf af = ldfrag(&pb[(size_t)tr * Sc + kk + 8 * g], &pb[(size_t)tr * Sc + kk + 16 + 8 * g]);
    const unsigned short* vp = &Vt[vBase + kk];
    v16bf bf_ = ldfrag(vp + 8 * g, vp + 16 + 8 * g);
    a = wmma_bf16(af, bf_, a);
  }
  float* obuf = sc;  // reuse score LDS for cross-wave reduction [4][16][16]
  if (half == 1) {
#pragma unroll
    for (int r = 0; r < 8; r++) obuf[(nt * 16 + 8 * g + r) * 16 + tr] = a[r];
  }
  __syncthreads();
  if (half == 0) {
#pragma unroll
    for (int r = 0; r < 8; r++) {
      const float v = a[r] + obuf[(nt * 16 + 8 * g + r) * 16 + tr];
      const int srow = qt * 16 + 8 * g + r;
      Ob[((size_t)bb * Sc + srow) * Dc + hh * DHc + nt * 16 + tr] = f2bf(v);
    }
  }
}

// ---------------- host orchestration ----------------
extern "C" void kernel_launch(void* const* d_in, const int* in_sizes, int n_in,
                              void* d_out, int out_size, void* d_ws, size_t ws_size,
                              hipStream_t stream) {
  (void)in_sizes; (void)n_in; (void)out_size; (void)ws_size;
  const float* x    = (const float*)d_in[0];
  const float* mask = (const float*)d_in[1];
  const float* ln1w = (const float*)d_in[2];
  const float* ln1b = (const float*)d_in[3];
  const float* inw  = (const float*)d_in[4];
  const float* inb  = (const float*)d_in[5];
  const float* outw = (const float*)d_in[6];
  const float* outb = (const float*)d_in[7];
  const float* ln2w = (const float*)d_in[8];
  const float* ln2b = (const float*)d_in[9];
  const float* f1w  = (const float*)d_in[10];
  const float* f1b  = (const float*)d_in[11];
  const float* f2w  = (const float*)d_in[12];
  const float* f2b  = (const float*)d_in[13];
  const float* nw   = (const float*)d_in[14];
  const float* nb   = (const float*)d_in[15];

  char* ws = (char*)d_ws;
  size_t off = 0;
  float* h = (float*)(ws + off);               off += (size_t)NT * Dc * 4;
  unsigned short* ybf  = (unsigned short*)(ws + off); off += (size_t)NT * Dc * 2;
  unsigned short* qb   = (unsigned short*)(ws + off); off += (size_t)NT * Dc * 2;
  unsigned short* kb   = (unsigned short*)(ws + off); off += (size_t)NT * Dc * 2;
  unsigned short* vb   = (unsigned short*)(ws + off); off += (size_t)NT * Dc * 2;
  unsigned short* obf  = (unsigned short*)(ws + off); off += (size_t)NT * Dc * 2;
  unsigned short* mid  = (unsigned short*)(ws + off); off += (size_t)NT * Fc * 2;
  unsigned short* w_in = (unsigned short*)(ws + off); off += (size_t)Lc * Dc * 3 * Dc * 2;
  unsigned short* w_ou = (unsigned short*)(ws + off); off += (size_t)Lc * Dc * Dc * 2;
  unsigned short* w_f1 = (unsigned short*)(ws + off); off += (size_t)Lc * Dc * Fc * 2;
  unsigned short* w_f2 = (unsigned short*)(ws + off); off += (size_t)Lc * Fc * Dc * 2;

  // weights converted bf16 AND transposed to [N][K] so GEMM staging is all-b128
  cvt_t_kernel<<<2048, 256, 0, stream>>>(inw,  w_in, Dc, 3 * Dc, (size_t)Lc * Dc * 3 * Dc);
  cvt_t_kernel<<<2048, 256, 0, stream>>>(outw, w_ou, Dc, Dc,     (size_t)Lc * Dc * Dc);
  cvt_t_kernel<<<2048, 256, 0, stream>>>(f1w,  w_f1, Dc, Fc,     (size_t)Lc * Dc * Fc);
  cvt_t_kernel<<<2048, 256, 0, stream>>>(f2w,  w_f2, Fc, Dc,     (size_t)Lc * Fc * Dc);
  hipMemcpyAsync(h, x, (size_t)NT * Dc * 4, hipMemcpyDeviceToDevice, stream);

  const size_t attnSmem = (size_t)16 * Sc * 4 + (size_t)16 * Sc * 2 + (size_t)16 * DHc * 2;
  hipFuncSetAttribute((const void*)attn_kernel, hipFuncAttributeMaxDynamicSharedMemorySize,
                      (int)attnSmem);

  for (int l = 0; l < Lc; l++) {
    ln_kernel<<<NT, 256, 0, stream>>>(h, ln1w + (size_t)l * Dc, ln1b + (size_t)l * Dc, nullptr, ybf);
    gemm_bf16<<<dim3(3 * Dc / 128, NT / 128), 256, 0, stream>>>(
        ybf, w_in + (size_t)l * Dc * 3 * Dc, inb + (size_t)l * 3 * Dc,
        nullptr, nullptr, nullptr, qb, kb, vb, NT, 3 * Dc, Dc, 0);
    attn_kernel<<<Bc * Hc * (Sc / 16), 256, attnSmem, stream>>>(qb, kb, vb, mask, obf);
    gemm_bf16<<<dim3(Dc / 128, NT / 128), 256, 0, stream>>>(
        obf, w_ou + (size_t)l * Dc * Dc, outb + (size_t)l * Dc,
        h, h, nullptr, nullptr, nullptr, nullptr, NT, Dc, Dc, 1);
    ln_kernel<<<NT, 256, 0, stream>>>(h, ln2w + (size_t)l * Dc, ln2b + (size_t)l * Dc, nullptr, ybf);
    gemm_bf16<<<dim3(Fc / 128, NT / 128), 256, 0, stream>>>(
        ybf, w_f1 + (size_t)l * Dc * Fc, f1b + (size_t)l * Fc,
        nullptr, nullptr, mid, nullptr, nullptr, nullptr, NT, Fc, Dc, 2);
    gemm_bf16<<<dim3(Dc / 128, NT / 128), 256, 0, stream>>>(
        mid, w_f2 + (size_t)l * Fc * Dc, f2b + (size_t)l * Dc,
        h, h, nullptr, nullptr, nullptr, nullptr, NT, Dc, Fc, 1);
  }
  ln_kernel<<<NT, 256, 0, stream>>>(h, nw, nb, (float*)d_out, nullptr);
}
